// LightS6_12498354831585
// MI455X (gfx1250) — matrix operand order repaired and checked by reading, not compiled, not run on hardware
//
#include <hip/hip_runtime.h>
#include <math.h>

// Problem sizes (fixed by the reference)
#define BQ      4
#define SQ      2048
#define DIMQ    1024
#define DSTATEQ 16
#define DTRANKQ 128
#define NPROJ   (DTRANKQ + 2 * DSTATEQ)   // 160

// Chunked-scan configuration: 8 chunks of 256 steps -> 8x wave parallelism
#define NCHUNK  8
#define CHUNKS  (SQ / NCHUNK)             // 256
#define DNQ     (DIMQ * DSTATEQ)          // 16384 per-(d,n) states per batch

typedef __attribute__((ext_vector_type(16))) __bf16 v16bf;
typedef __attribute__((ext_vector_type(8)))  float  v8f;

// ---------------------------------------------------------------------------
// Split-bf16 fragment builders (fp32 -> bf16_hi + bf16_lo, RNE each step).
// A-matrix (16x32, 16-bit): lane m = lane&15; K chunks {half*8 .. +7} and
// {half*8+16 .. +7}  (ISA 7.12.2 16-bit A layout).
// B-matrix (32x16, 16-bit): lane n = lane&15; K chunks {half*8 .. +7} and {half*8+16 .. +7}.
// ---------------------------------------------------------------------------
__device__ __forceinline__ void build_a_frag(const float* __restrict__ arow,
                                             int k, int half,
                                             v16bf& hi, v16bf& lo) {
    const float* p = arow + k + half * 8;
#pragma unroll
    for (int i = 0; i < 8; ++i) {
        float f = p[i];
        __bf16 h = (__bf16)f;
        hi[i] = h;
        lo[i] = (__bf16)(f - (float)h);
    }
    p += 16;
#pragma unroll
    for (int i = 0; i < 8; ++i) {
        float f = p[i];
        __bf16 h = (__bf16)f;
        hi[8 + i] = h;
        lo[8 + i] = (__bf16)(f - (float)h);
    }
}

__device__ __forceinline__ void build_b_frag(const float* __restrict__ brow,
                                             int k, int half,
                                             v16bf& hi, v16bf& lo) {
    const float* p = brow + k + half * 8;
#pragma unroll
    for (int i = 0; i < 16; ++i) {
        float f = p[i + (i >> 3) * 8];
        __bf16 h = (__bf16)f;
        hi[i] = h;
        lo[i] = (__bf16)(f - (float)h);
    }
}

// bf16x3 accumulation: acc += A_hi*B_hi + A_lo*B_hi + A_hi*B_lo  (f32 acc)
__device__ __forceinline__ v8f wmma3(v16bf ah, v16bf al, v16bf bh, v16bf bl, v8f acc) {
    acc = __builtin_amdgcn_wmma_f32_16x16x32_bf16(false, ah, false, bh, (short)0, acc, false, false);
    acc = __builtin_amdgcn_wmma_f32_16x16x32_bf16(false, al, false, bh, (short)0, acc, false, false);
    acc = __builtin_amdgcn_wmma_f32_16x16x32_bf16(false, ah, false, bl, (short)0, acc, false, false);
    return acc;
}

// ---------------------------------------------------------------------------
// Kernel 1: proj = x @ Wx^T.   A = x (8192 x 1024), B = Wx (160 x 1024, N-major)
// One wave per 16x16 output tile; 4 waves / block.
// ---------------------------------------------------------------------------
__global__ void s6_proj_gemm(const float* __restrict__ X,
                             const float* __restrict__ Wx,
                             float* __restrict__ proj) {
    const int lane  = threadIdx.x & 31;
    const int wave  = threadIdx.x >> 5;
    const int mtile = blockIdx.x * 4 + wave;
    const int ntile = blockIdx.y;
    const int half  = lane >> 4;
    const int lr    = lane & 15;

    const float* arow = X  + (size_t)(mtile * 16 + lr) * DIMQ;
    const float* brow = Wx + (size_t)(ntile * 16 + lr) * DIMQ;

    v8f acc = {};
    for (int k = 0; k < DIMQ; k += 32) {
        v16bf ah, al, bh, bl;
        build_a_frag(arow, k, half, ah, al);
        build_b_frag(brow, k, half, bh, bl);
        acc = wmma3(ah, al, bh, bl, acc);
    }

#pragma unroll
    for (int r = 0; r < 8; ++r) {
        int m = mtile * 16 + r + half * 8;
        proj[(size_t)m * NPROJ + ntile * 16 + lr] = acc[r];
    }
}

// ---------------------------------------------------------------------------
// Kernel 2: dt = silu(dt_raw @ Wdt^T + bdt).
// A = proj[:, 0:128] (lda = NPROJ), B = Wdt (1024 x 128, N-major), K = 128.
// ---------------------------------------------------------------------------
__global__ void s6_dt_gemm(const float* __restrict__ proj,
                           const float* __restrict__ Wdt,
                           const float* __restrict__ bdt,
                           float* __restrict__ dtb) {
    const int lane  = threadIdx.x & 31;
    const int wave  = threadIdx.x >> 5;
    const int mtile = blockIdx.x * 4 + wave;
    const int ntile = blockIdx.y;
    const int half  = lane >> 4;
    const int lr    = lane & 15;

    const float* arow = proj + (size_t)(mtile * 16 + lr) * NPROJ;
    const float* brow = Wdt  + (size_t)(ntile * 16 + lr) * DTRANKQ;

    v8f acc = {};
#pragma unroll
    for (int k = 0; k < DTRANKQ; k += 32) {
        v16bf ah, al, bh, bl;
        build_a_frag(arow, k, half, ah, al);
        build_b_frag(brow, k, half, bh, bl);
        acc = wmma3(ah, al, bh, bl, acc);
    }

    const int   d    = ntile * 16 + lr;
    const float bias = bdt[d];
#pragma unroll
    for (int r = 0; r < 8; ++r) {
        int   m = mtile * 16 + r + half * 8;
        float v = acc[r] + bias;
        float sig = 1.0f / (1.0f + __expf(-v));        // silu
        dtb[(size_t)m * DIMQ + d] = v * sig;
    }
}

// ---------------------------------------------------------------------------
// Scan, chunked two-pass (diagonal linear recurrence is associative):
//   Pass 1: per chunk, decay product cp = prod(dA) and zero-init local end
//           state. 8x wave parallelism, minimal chain (exp+fma only).
//   Pass 2: tiny sequential combine over chunks -> per-chunk initial states.
//   Pass 3: per chunk, full scan seeded with correct initial state, produces
//           y with 16-lane xor reduction.
// Lane map: n = lane&15 (DSTATE), dsub = lane>>4 (2 d-channels / wave),
// block = 8 waves = 16 d channels; grid (DIM/16, B*NCHUNK).
// ---------------------------------------------------------------------------
__global__ void s6_scan_pass1(const float* __restrict__ X,
                              const float* __restrict__ proj,
                              const float* __restrict__ dtb,
                              const float* __restrict__ A_log,
                              float* __restrict__ carry,
                              float* __restrict__ endl) {
    const int lane  = threadIdx.x & 31;
    const int wave  = threadIdx.x >> 5;
    const int n     = lane & 15;
    const int dsub  = lane >> 4;
    const int d     = blockIdx.x * 16 + wave * 2 + dsub;
    const int bc    = blockIdx.y;            // b * NCHUNK + chunk
    const int b     = bc >> 3;               // NCHUNK == 8
    const int chunk = bc & (NCHUNK - 1);

    const float a_ln = -__expf(A_log[(size_t)d * DSTATEQ + n]);

    float state = 0.0f;
    float cp    = 1.0f;
    const size_t bs0 = (size_t)b * SQ + (size_t)chunk * CHUNKS;

#pragma unroll 2
    for (int s = 0; s < CHUNKS; ++s) {
        const size_t bs = bs0 + s;
        float bv  = proj[bs * NPROJ + DTRANKQ + n];
        float dtv = dtb[bs * DIMQ + d];
        float xv  = X[bs * DIMQ + d];

        if (s + 8 < CHUNKS) {
            __builtin_prefetch(dtb + (bs + 8) * DIMQ + d, 0, 0);
            __builtin_prefetch(X   + (bs + 8) * DIMQ + d, 0, 0);
        }

        float dA = __expf(dtv * a_ln);
        cp *= dA;
        state = __builtin_fmaf(dA, state, dtv * bv * xv);
    }

    const size_t idx = (size_t)bc * DNQ + (size_t)d * DSTATEQ + n;
    carry[idx] = cp;
    endl[idx]  = state;
}

// init[b,chunk,dn]: state entering each chunk. One thread per (b,dn).
__global__ void s6_scan_combine(const float* __restrict__ carry,
                                const float* __restrict__ endl,
                                float* __restrict__ init) {
    const int gid = blockIdx.x * blockDim.x + threadIdx.x;   // 0 .. B*DNQ-1
    const int b   = gid / DNQ;
    const int dn  = gid - b * DNQ;
    const size_t base = (size_t)b * NCHUNK * DNQ + dn;

    float s = 0.0f;
#pragma unroll
    for (int c = 0; c < NCHUNK; ++c) {
        const size_t idx = base + (size_t)c * DNQ;
        init[idx] = s;
        s = __builtin_fmaf(carry[idx], s, endl[idx]);
    }
}

__global__ void s6_scan_pass3(const float* __restrict__ X,
                              const float* __restrict__ proj,
                              const float* __restrict__ dtb,
                              const float* __restrict__ A_log,
                              const float* __restrict__ Dv,
                              const float* __restrict__ init,
                              float* __restrict__ Y) {
    const int lane  = threadIdx.x & 31;
    const int wave  = threadIdx.x >> 5;
    const int n     = lane & 15;
    const int dsub  = lane >> 4;
    const int d     = blockIdx.x * 16 + wave * 2 + dsub;
    const int bc    = blockIdx.y;
    const int b     = bc >> 3;
    const int chunk = bc & (NCHUNK - 1);

    const float a_ln = -__expf(A_log[(size_t)d * DSTATEQ + n]);
    const float Dd   = Dv[d];

    float state = init[(size_t)bc * DNQ + (size_t)d * DSTATEQ + n];
    const size_t bs0 = (size_t)b * SQ + (size_t)chunk * CHUNKS;

#pragma unroll 2
    for (int s = 0; s < CHUNKS; ++s) {
        const size_t bs = bs0 + s;
        const float* pr = proj + bs * NPROJ;

        float bv  = pr[DTRANKQ + n];                 // B state input
        float cv  = pr[DTRANKQ + DSTATEQ + n];       // C state output
        float dtv = dtb[bs * DIMQ + d];
        float xv  = X[bs * DIMQ + d];

        if (s + 8 < CHUNKS) {
            __builtin_prefetch(dtb + (bs + 8) * DIMQ + d, 0, 0);
            __builtin_prefetch(X   + (bs + 8) * DIMQ + d, 0, 0);
        }

        float dA = __expf(dtv * a_ln);
        state = __builtin_fmaf(dA, state, dtv * bv * xv);

        float part = state * cv;                     // off the state chain
        part += __shfl_xor(part, 8, 32);
        part += __shfl_xor(part, 4, 32);
        part += __shfl_xor(part, 2, 32);
        part += __shfl_xor(part, 1, 32);

        if (n == 0) Y[bs * DIMQ + d] = __builtin_fmaf(xv, Dd, part);
    }
}

// ---------------------------------------------------------------------------
extern "C" void kernel_launch(void* const* d_in, const int* in_sizes, int n_in,
                              void* d_out, int out_size, void* d_ws, size_t ws_size,
                              hipStream_t stream) {
    const float* x     = (const float*)d_in[0];
    const float* Wx    = (const float*)d_in[1];
    const float* Wdt   = (const float*)d_in[2];
    const float* bdt   = (const float*)d_in[3];
    const float* A_log = (const float*)d_in[4];
    const float* Dv    = (const float*)d_in[5];
    float* y = (float*)d_out;

    const size_t proj_elems  = (size_t)BQ * SQ * NPROJ;        // 1,310,720 f32
    const size_t dt_elems    = (size_t)BQ * SQ * DIMQ;         // 8,388,608 f32
    const size_t chunk_elems = (size_t)BQ * NCHUNK * DNQ;      //   524,288 f32
    if (ws_size < (proj_elems + dt_elems + 3 * chunk_elems) * sizeof(float)) return;

    float* proj  = (float*)d_ws;
    float* dtb   = proj  + proj_elems;
    float* carry = dtb   + dt_elems;
    float* endl  = carry + chunk_elems;
    float* init  = endl  + chunk_elems;

    const int mtiles = (BQ * SQ) / 16;                         // 512

    dim3 gblk(128, 1, 1);                                      // 4 waves / block
    s6_proj_gemm<<<dim3(mtiles / 4, NPROJ / 16, 1), gblk, 0, stream>>>(x, Wx, proj);
    s6_dt_gemm  <<<dim3(mtiles / 4, DIMQ  / 16, 1), gblk, 0, stream>>>(proj, Wdt, bdt, dtb);

    dim3 sblk(256, 1, 1);                                      // 8 waves / block
    dim3 sgrd(DIMQ / 16, BQ * NCHUNK, 1);                      // (64, 32)
    s6_scan_pass1 <<<sgrd, sblk, 0, stream>>>(x, proj, dtb, A_log, carry, endl);
    s6_scan_combine<<<dim3((BQ * DNQ) / 256, 1, 1), sblk, 0, stream>>>(carry, endl, init);
    s6_scan_pass3 <<<sgrd, sblk, 0, stream>>>(x, proj, dtb, A_log, Dv, init, y);
}
